// CausalSelfAttention_23536420782348
// MI455X (gfx1250) — matrix operand order repaired and checked by reading, not compiled, not run on hardware
//
#include <hip/hip_runtime.h>
#include <hip/hip_bf16.h>
#include <math.h>

typedef __attribute__((ext_vector_type(16))) _Float16 v16h;
typedef __attribute__((ext_vector_type(8)))  float    v8f;

#define EMBED  1024
#define E3     3072
#define NHEAD  16
#define HDIM   64
#define SEQ    77
#define BATCH  1024
#define NROWS  (BATCH * SEQ)     /* 78848, multiple of 64 */
#define RPAD   (NROWS + 32)
#define SCALE_F 0.125f           /* 1/sqrt(64) */

/* ---------- fragment loaders (contiguous k order, wave32) ---------- */

/* A matrix 16x32 f16: lane L: m = L&15, kbase = (L>>4)*16.
   halves h -> K = kbase+h (contiguous). */
static __device__ __forceinline__ v16h frag_a(const _Float16* __restrict__ base, int ld) {
    int lane = threadIdx.x & 31;
    int m  = lane & 15;
    int kb = (lane >> 4) << 4;
    const _Float16* p = base + (size_t)m * ld + kb;
    v16h r;
    *((uint4*)&r)     = *(const uint4*)(p);
    *((uint4*)&r + 1) = *(const uint4*)(p + 8);
    return r;
}

/* B matrix 32x16 f16 from a [N,K] row-major buffer (i.e. BT):
   lane L: n = L&15, koff = (L>>4)*16, halves h -> K = koff+h (contiguous). */
static __device__ __forceinline__ v16h frag_b_T(const _Float16* __restrict__ baseT, int ld) {
    int lane = threadIdx.x & 31;
    int n  = lane & 15;
    int ko = (lane >> 4) << 4;
    const _Float16* p = baseT + (size_t)n * ld + ko;
    v16h r;
    *((uint4*)&r)     = *(const uint4*)(p);
    *((uint4*)&r + 1) = *(const uint4*)(p + 8);
    return r;
}

/* B matrix 32x16 f16 from a [K,N] row-major buffer (strided gather). */
static __device__ __forceinline__ v16h frag_b_KN(const _Float16* __restrict__ base, int ld) {
    int lane = threadIdx.x & 31;
    int n  = lane & 15;
    int ko = (lane >> 4) << 4;
    v16h r;
#pragma unroll
    for (int h = 0; h < 16; ++h)
        r[h] = base[(size_t)(ko + h) * ld + n];
    return r;
}

static __device__ __forceinline__ v8f wmma_f16(v16h a, v16h b, v8f c) {
    return __builtin_amdgcn_wmma_f32_16x16x32_f16(false, a, false, b, (short)0, c, false, false);
}

/* ---------- pre-pass kernels ---------- */

__global__ void k_f32_to_f16(const float* __restrict__ s, _Float16* __restrict__ d, size_t n) {
    size_t i = (size_t)blockIdx.x * blockDim.x + threadIdx.x;
    if (i < n) d[i] = (_Float16)s[i];
}

/* W [K,N] f32 -> WT [N,K] f16 */
__global__ void k_transpose_h(const float* __restrict__ W, _Float16* __restrict__ WT,
                              int K, int N) {
    size_t i = (size_t)blockIdx.x * blockDim.x + threadIdx.x;
    if (i >= (size_t)K * N) return;
    int n = (int)(i / K);
    int k = (int)(i % K);
    WT[i] = (_Float16)W[(size_t)k * N + n];
}

__global__ void k_zero_f16(_Float16* __restrict__ p, size_t n) {
    size_t i = (size_t)blockIdx.x * blockDim.x + threadIdx.x;
    if (i < n) p[i] = (_Float16)0.f;
}

/* ---------- GEMM: C[M,N] = A[M,K] (f16) x BT[N,K] (f16) + bias ----------
   Wave tile 64x64: 16 WMMAs per K-step of 32, 16 B128 loads per K-step
   (each A and B fragment reused 4x). Exact grids -> EXEC all-ones. */

template <bool OUTF16>
__global__ __launch_bounds__(256) void k_gemm(const _Float16* __restrict__ A,
                                              const _Float16* __restrict__ BT,
                                              const float* __restrict__ bias,
                                              void* __restrict__ Cout,
                                              int K, int ldc, int nblk) {
    int wid  = (int)((blockIdx.x * blockDim.x + threadIdx.x) >> 5);
    int lane = threadIdx.x & 31;
    int mb = wid / nblk;    /* 64-row block */
    int nb = wid % nblk;    /* 64-col block */

    v8f acc[4][4];
#pragma unroll
    for (int mt = 0; mt < 4; ++mt)
#pragma unroll
        for (int nt = 0; nt < 4; ++nt) acc[mt][nt] = {};

    const _Float16* Am[4];
#pragma unroll
    for (int mt = 0; mt < 4; ++mt)
        Am[mt] = A + ((size_t)mb * 64 + (size_t)mt * 16) * K;
    const _Float16* Bb = BT + (size_t)nb * 64 * K;

    for (int k = 0; k < K; k += 32) {
        v16h a[4];
#pragma unroll
        for (int mt = 0; mt < 4; ++mt) a[mt] = frag_a(Am[mt] + k, K);
#pragma unroll
        for (int nt = 0; nt < 4; ++nt) {
            v16h b = frag_b_T(Bb + (size_t)nt * 16 * K + k, K);
#pragma unroll
            for (int mt = 0; mt < 4; ++mt)
                acc[mt][nt] = wmma_f16(a[mt], b, acc[mt][nt]);
        }
    }

    int ncol   = lane & 15;
    int rowoff = (lane >> 4) << 3;
#pragma unroll
    for (int nt = 0; nt < 4; ++nt) {
        int col = nb * 64 + nt * 16 + ncol;
        float bv = bias[col];
#pragma unroll
        for (int mt = 0; mt < 4; ++mt) {
#pragma unroll
            for (int r = 0; r < 8; ++r) {
                int row = mb * 64 + mt * 16 + r + rowoff;
                float v = acc[mt][nt][r] + bv;
                if (OUTF16)
                    ((_Float16*)Cout)[(size_t)row * ldc + col] = (_Float16)v;
                else
                    ((float*)Cout)[(size_t)row * ldc + col] = v;
            }
        }
    }
}

/* ---------- attention: one wave per (b, h, q-tile of 16 rows) ---------- */

__global__ __launch_bounds__(256) void k_attn(const _Float16* __restrict__ qkv, /* [RPAD,3072] */
                                              _Float16* __restrict__ yh) {      /* [NROWS,1024] */
    __shared__ _Float16 smP[8][16 * 96];

    int w    = threadIdx.x >> 5;
    int lane = threadIdx.x & 31;
    int gw   = blockIdx.x * 8 + w;
    int qt = gw % 5;
    int h  = (gw / 5) & 15;
    int b  = gw / 80;

    const _Float16* base = qkv + (size_t)b * SEQ * E3;
    const _Float16* qb = base + h * HDIM + (size_t)qt * 16 * E3;
    const _Float16* kb = base + EMBED + h * HDIM;
    const _Float16* vb = base + 2 * EMBED + h * HDIM;

    v16h qa0 = frag_a(qb, E3);
    v16h qa1 = frag_a(qb + 32, E3);

    /* scores: 5 tiles of 16x16, contracting over D=64 */
    v8f s[5];
#pragma unroll
    for (int nt = 0; nt < 5; ++nt) {
        v8f c = {};
        v16h b0 = frag_b_T(kb + (size_t)nt * 16 * E3, E3);
        c = wmma_f16(qa0, b0, c);
        v16h b1 = frag_b_T(kb + (size_t)nt * 16 * E3 + 32, E3);
        c = wmma_f16(qa1, b1, c);
        s[nt] = c;
    }

    int ncol   = lane & 15;
    int rowoff = (lane >> 4) << 3;

    /* scale + causal mask */
#pragma unroll
    for (int nt = 0; nt < 5; ++nt) {
#pragma unroll
        for (int r = 0; r < 8; ++r) {
            int i = qt * 16 + r + rowoff;
            int j = nt * 16 + ncol;
            bool valid = (j <= i) && (j < SEQ);
            s[nt][r] = valid ? s[nt][r] * SCALE_F : -1.0e30f;
        }
    }

    /* row softmax (rows live across 16-lane groups of each C fragment) */
    float rmax[8], rsum[8];
#pragma unroll
    for (int r = 0; r < 8; ++r) {
        float m = s[0][r];
#pragma unroll
        for (int nt = 1; nt < 5; ++nt) m = fmaxf(m, s[nt][r]);
        for (int d = 1; d < 16; d <<= 1) m = fmaxf(m, __shfl_xor(m, d, 32));
        rmax[r] = m;
    }
#pragma unroll
    for (int r = 0; r < 8; ++r) {
        float t = 0.f;
#pragma unroll
        for (int nt = 0; nt < 5; ++nt) {
            float e = __expf(s[nt][r] - rmax[r]);
            s[nt][r] = e;
            t += e;
        }
        for (int d = 1; d < 16; d <<= 1) t += __shfl_xor(t, d, 32);
        rsum[r] = t;
    }

    /* P (f16, normalized) -> LDS [16 x 96], zero pad cols 80..95 */
    _Float16* sp = smP[w];
#pragma unroll
    for (int r = 0; r < 8; ++r) {
        float inv = 1.0f / rsum[r];
        int rr = r + rowoff;
#pragma unroll
        for (int nt = 0; nt < 5; ++nt)
            sp[rr * 96 + nt * 16 + ncol] = (_Float16)(s[nt][r] * inv);
        sp[rr * 96 + 80 + ncol] = (_Float16)0.f;
    }

    /* O = P x V : contraction over s_kv (padded to 96), 4 output d-tiles */
    v8f o[4];
#pragma unroll
    for (int nt = 0; nt < 4; ++nt) o[nt] = {};
#pragma unroll
    for (int kk = 0; kk < 3; ++kk) {
        v16h pa = frag_a(sp + kk * 32, 96);
#pragma unroll
        for (int nt = 0; nt < 4; ++nt) {
            v16h bv = frag_b_KN(vb + (size_t)kk * 32 * E3 + nt * 16, E3);
            o[nt] = wmma_f16(pa, bv, o[nt]);
        }
    }

    /* store y (f16, [B,S,E] layout) */
#pragma unroll
    for (int nt = 0; nt < 4; ++nt) {
#pragma unroll
        for (int r = 0; r < 8; ++r) {
            int sl = qt * 16 + r + rowoff;
            if (sl < SEQ)
                yh[(size_t)(b * SEQ + sl) * EMBED + h * HDIM + nt * 16 + ncol] = (_Float16)o[nt][r];
        }
    }
}

/* ---------- launch ---------- */

extern "C" void kernel_launch(void* const* d_in, const int* in_sizes, int n_in,
                              void* d_out, int out_size, void* d_ws, size_t ws_size,
                              hipStream_t stream) {
    const float* x      = (const float*)d_in[0];
    const float* W_attn = (const float*)d_in[1];
    const float* b_attn = (const float*)d_in[2];
    const float* W_proj = (const float*)d_in[3];
    const float* b_proj = (const float*)d_in[4];
    float* out = (float*)d_out;

    _Float16* ws = (_Float16*)d_ws;
    size_t off = 0;
    _Float16* x_h  = ws + off; off += (size_t)NROWS * EMBED;   /* x in f16 */
    _Float16* waT  = ws + off; off += (size_t)E3 * EMBED;      /* W_attn^T f16 [3072,1024] */
    _Float16* wpT  = ws + off; off += (size_t)EMBED * EMBED;   /* W_proj^T f16 [1024,1024] */
    _Float16* qkvh = ws + off; off += (size_t)RPAD * E3;       /* qkv f16 (+pad rows) */
    _Float16* y_h  = ws + off; off += (size_t)NROWS * EMBED;   /* attention output f16 */
    (void)ws_size; (void)in_sizes; (void)n_in; (void)out_size;

    /* pre-pass: conversions / transposes / pad-zero */
    {
        size_t n = (size_t)NROWS * EMBED;
        k_f32_to_f16<<<(unsigned)((n + 255) / 256), 256, 0, stream>>>(x, x_h, n);
    }
    {
        size_t n = (size_t)EMBED * E3;
        k_transpose_h<<<(unsigned)((n + 255) / 256), 256, 0, stream>>>(W_attn, waT, EMBED, E3);
    }
    {
        size_t n = (size_t)EMBED * EMBED;
        k_transpose_h<<<(unsigned)((n + 255) / 256), 256, 0, stream>>>(W_proj, wpT, EMBED, EMBED);
    }
    {
        size_t n = (size_t)32 * E3;
        k_zero_f16<<<(unsigned)((n + 255) / 256), 256, 0, stream>>>(qkvh + (size_t)NROWS * E3, n);
    }

    /* qkv = x_h @ W_attn + b_attn  -> f16 [NROWS,3072]
       waves = (NROWS/64) * (3072/64) = 1232*48 = 59136 -> 7392 blocks of 256 */
    k_gemm<true><<<7392, 256, 0, stream>>>(x_h, waT, b_attn, (void*)qkvh, EMBED, E3, 48);

    /* attention: 1024*16*5 = 81920 waves -> 10240 blocks */
    k_attn<<<10240, 256, 0, stream>>>(qkvh, y_h);

    /* out = y_h @ W_proj + b_proj -> f32 [NROWS,1024]
       waves = (NROWS/64) * (1024/64) = 1232*16 = 19712 -> 2464 blocks */
    k_gemm<false><<<2464, 256, 0, stream>>>(y_h, wpT, b_proj, (void*)out, EMBED, EMBED, 16);
}
